// LongRangeGPT_67362267071078
// MI455X (gfx1250) — matrix operand order repaired and checked by reading, not compiled, not run on hardware
//
#include <hip/hip_runtime.h>

// ---------------------------------------------------------------------------
// Types for CDNA5 WMMA (wave32): v_wmma_f32_16x16x32_bf16
// ---------------------------------------------------------------------------
typedef __bf16 v16bf __attribute__((ext_vector_type(16)));
typedef float  v8f   __attribute__((ext_vector_type(8)));

union U16 {            // 16 bf16 operand fragment (8 VGPRs)
    v16bf v;
    uint4 q[2];
    unsigned short s[16];
};
union F8 {             // 16x16 f32 accumulator fragment (8 VGPRs)
    v8f   v;
    float f[8];
};

// native hardware convert (v_cvt_*_bf16), round-to-nearest-even
static __device__ __forceinline__ unsigned short f2bf(float f) {
    __bf16 h = (__bf16)f;
    return __builtin_bit_cast(unsigned short, h);
}

// ---------------------------------------------------------------------------
// Embedding: x[b,t,:] = wte[ids[b,t],:] + wpe[t,:]
// ---------------------------------------------------------------------------
__global__ void embed_kernel(const int* __restrict__ ids,
                             const float* __restrict__ wte,
                             const float* __restrict__ wpe,
                             float* __restrict__ x) {
    int i = blockIdx.x * blockDim.x + threadIdx.x;   // over 2048*768
    if (i >= 2048 * 768) return;
    int row = i / 768, d = i - row * 768;
    int t = row & 1023;
    x[i] = wte[(size_t)ids[row] * 768 + d] + wpe[(size_t)t * 768 + d];
}

// ---------------------------------------------------------------------------
// LayerNorm over D=768, emits bf16 (GEMM A operand). One block per row.
// ---------------------------------------------------------------------------
__global__ __launch_bounds__(256)
void ln_bf16_kernel(const float* __restrict__ x, const float* __restrict__ g,
                    const float* __restrict__ be, unsigned short* __restrict__ out) {
    __shared__ float red[256];
    const int row = blockIdx.x, tid = threadIdx.x;
    const float* xr = x + (size_t)row * 768;
    float v0 = xr[tid], v1 = xr[tid + 256], v2 = xr[tid + 512];
    red[tid] = v0 + v1 + v2;
    __syncthreads();
    for (int d = 128; d; d >>= 1) { if (tid < d) red[tid] += red[tid + d]; __syncthreads(); }
    float mu = red[0] * (1.0f / 768.0f);
    __syncthreads();
    float d0 = v0 - mu, d1 = v1 - mu, d2 = v2 - mu;
    red[tid] = d0 * d0 + d1 * d1 + d2 * d2;
    __syncthreads();
    for (int d = 128; d; d >>= 1) { if (tid < d) red[tid] += red[tid + d]; __syncthreads(); }
    float rs = rsqrtf(red[0] * (1.0f / 768.0f) + 1e-5f);
    unsigned short* orow = out + (size_t)row * 768;
    orow[tid]       = f2bf(d0 * rs * g[tid]       + be[tid]);
    orow[tid + 256] = f2bf(d1 * rs * g[tid + 256] + be[tid + 256]);
    orow[tid + 512] = f2bf(d2 * rs * g[tid + 512] + be[tid + 512]);
}

// ---------------------------------------------------------------------------
// WMMA GEMM: out[M,N] = A_bf16[M,K] @ W_f32[K,N] (+bias)(+gelu)(+resid)
// Block = 256 threads = 8 waves; tile 128x128; K-step 32 (one WMMA).
// Register-pipelined: tile k+32 is fetched into VGPRs while WMMAs run.
// W staging is fully coalesced (flat-index mapping) and guard-free when the
// whole 128-column tile is in-bounds (block-uniform fast path).
// transOutB=1: bf16 output written transposed as outB[col*M + row]
// (feature-major, used so attention can read V with contiguous b128 loads).
// M must be a multiple of 128 (true for all calls: 2048 / 4096).
// ---------------------------------------------------------------------------
__global__ __launch_bounds__(256)
void gemm_bf16_kernel(const unsigned short* __restrict__ A,
                      const float* __restrict__ W,
                      const float* __restrict__ bias,
                      const float* __restrict__ resid,
                      float* __restrict__ outF,
                      unsigned short* __restrict__ outB,
                      int M, int N, int K, int fuseGelu, int transOutB) {
    __shared__ __align__(16) unsigned short As[128 * 32];   // [m][k]  8 KB
    __shared__ __align__(16) unsigned short Bs[128 * 40];   // [n][k] 10 KB (pad 40)
    const int tid  = threadIdx.x;
    const int m0   = blockIdx.y * 128, n0 = blockIdx.x * 128;
    const int lane = tid & 31, w = tid >> 5;
    const int half = lane >> 4, lm = lane & 15;
    const int wm   = (w & 3) * 32, wn = (w >> 2) * 64;
    const bool fullN = (n0 + 128 <= N);

    const int ar = tid >> 1, ac = (tid & 1) * 16;   // A tile: 2 thr/row, 16 bf16
    const int wrow0 = tid >> 7;                      // W tile flat map: row=idx>>7
    const int wcol  = tid & 127;                     //                  col=idx&127

    F8 acc[2][4] = {};
    uint4 a0, a1;
    float wv[16];

    auto fetch = [&](int k0) {
        const uint4* asrc = (const uint4*)(A + (size_t)(m0 + ar) * K + k0 + ac);
        a0 = asrc[0]; a1 = asrc[1];
        const float* wbase = W + (size_t)k0 * N + n0;
        if (fullN) {
#pragma unroll
            for (int r = 0; r < 16; ++r)
                wv[r] = wbase[(size_t)(wrow0 + r * 2) * N + wcol];
        } else {
            const bool ok = (n0 + wcol) < N;
#pragma unroll
            for (int r = 0; r < 16; ++r)
                wv[r] = ok ? wbase[(size_t)(wrow0 + r * 2) * N + wcol] : 0.0f;
        }
    };

    fetch(0);
    for (int k0 = 0; k0 < K; k0 += 32) {
        // commit staged registers to LDS
        {
            uint4* dst = (uint4*)&As[ar * 32 + ac];
            dst[0] = a0; dst[1] = a1;
        }
#pragma unroll
        for (int r = 0; r < 16; ++r)
            Bs[wcol * 40 + wrow0 + r * 2] = f2bf(wv[r]);
        __syncthreads();

        // prefetch next tile into registers (overlaps with WMMA below)
        if (k0 + 32 < K) fetch(k0 + 32);

        U16 fa[2], fb[4];
#pragma unroll
        for (int mi = 0; mi < 2; ++mi) {
            int row = wm + mi * 16 + lm;
            fa[mi].q[0] = *(const uint4*)&As[row * 32 + half * 8];        // j=0..7
            fa[mi].q[1] = *(const uint4*)&As[row * 32 + 16 + half * 8];   // j=8..15
        }
#pragma unroll
        for (int ni = 0; ni < 4; ++ni) {
            int nn = wn + ni * 16 + lm;
            fb[ni].q[0] = *(const uint4*)&Bs[nn * 40 + half * 8];         // j=0..7
            fb[ni].q[1] = *(const uint4*)&Bs[nn * 40 + 16 + half * 8];    // j=8..15
        }
#pragma unroll
        for (int mi = 0; mi < 2; ++mi)
#pragma unroll
            for (int ni = 0; ni < 4; ++ni)
                acc[mi][ni].v = __builtin_amdgcn_wmma_f32_16x16x32_bf16(
                    false, fa[mi].v, false, fb[ni].v, (short)0, acc[mi][ni].v,
                    false, false);
        __syncthreads();
    }

    // epilogue
#pragma unroll
    for (int mi = 0; mi < 2; ++mi)
#pragma unroll
        for (int ni = 0; ni < 4; ++ni) {
            const int col = n0 + wn + ni * 16 + lm;
            if (col < N) {
                const float bv = bias ? bias[col] : 0.0f;
#pragma unroll
                for (int r = 0; r < 8; ++r) {
                    int row = m0 + wm + mi * 16 + half * 8 + r;
                    float v = acc[mi][ni].f[r] + bv;
                    if (fuseGelu) v = 0.5f * v * (1.0f + erff(v * 0.70710678118654752f));
                    if (resid) v += resid[(size_t)row * N + col];
                    if (outF) outF[(size_t)row * N + col] = v;
                    if (outB) {
                        if (transOutB) outB[(size_t)col * M + row] = f2bf(v);
                        else           outB[(size_t)row * N + col] = f2bf(v);
                    }
                }
            }
        }
}

// ---------------------------------------------------------------------------
// Flash attention (online softmax). One wave per (b, h, 16-row tile).
// Q/K are bf16 [B*T, 768]; V is bf16 TRANSPOSED [768, B*Tkv] (vld = B*Tkv),
// so Q, K and V fragments are all direct aligned b128 loads.
// Writes bf16 output (feeds the Wo projection GEMM).
// kind: 0 = full, 1 = causal with diag offset (Tkv - Lq), 2 = doublecausal.
// ---------------------------------------------------------------------------
__global__ __launch_bounds__(32)
void flash_attn_kernel(const unsigned short* __restrict__ Q,
                       const unsigned short* __restrict__ Km,
                       const unsigned short* __restrict__ VT,
                       unsigned short* __restrict__ Obf,
                       int Lq, int Tkv, int vld, int kind) {
    __shared__ __align__(16) unsigned short Plds[16 * 32];
    const int lane = threadIdx.x;
    const int half = lane >> 4, m = lane & 15;
    const int l0 = blockIdx.x * 16;
    const int h  = blockIdx.y;
    const int b  = blockIdx.z;
    const unsigned short* qb  = Q  + (size_t)b * Lq  * 768 + h * 64;
    const unsigned short* kb  = Km + (size_t)b * Tkv * 768 + h * 64;
    const unsigned short* vbT = VT + (size_t)(h * 64) * vld + (size_t)b * Tkv;

    // Q fragments (16x64 contraction = 2 k-steps of 32): contiguous 16B loads
    U16 aq[2];
    {
        const unsigned short* qr = qb + (size_t)(l0 + m) * 768;
#pragma unroll
        for (int ks = 0; ks < 2; ++ks) {
            aq[ks].q[0] = *(const uint4*)(qr + ks * 32 + half * 8);
            aq[ks].q[1] = *(const uint4*)(qr + ks * 32 + 16 + half * 8);
        }
    }

    F8 O[4] = {};
    float mrow[8], lrow[8];
#pragma unroll
    for (int r = 0; r < 8; ++r) { mrow[r] = -3.0e38f; lrow[r] = 0.0f; }
    const int off = Tkv - Lq, halfT = Tkv >> 1;

    for (int t0 = 0; t0 < Tkv; t0 += 32) {
        if (kind == 1 && t0 > l0 + 15 + off) break;
        if (kind == 2) {
            if (t0 >= halfT && (t0 - halfT) > l0 + 15) break;      // past 2nd diag
            if (t0 + 31 < halfT && t0 > l0 + 15) continue;         // dead 1st-half tile
        }
        // S = Q @ K^T for 32 key columns (two 16-col tiles)
        F8 S[2] = {};
#pragma unroll
        for (int s = 0; s < 2; ++s) {
            const unsigned short* kr = kb + (size_t)(t0 + s * 16 + m) * 768;
#pragma unroll
            for (int ks = 0; ks < 2; ++ks) {
                U16 bk_;
                bk_.q[0] = *(const uint4*)(kr + ks * 32 + half * 8);
                bk_.q[1] = *(const uint4*)(kr + ks * 32 + 16 + half * 8);
                S[s].v = __builtin_amdgcn_wmma_f32_16x16x32_bf16(
                    false, aq[ks].v, false, bk_.v, (short)0, S[s].v, false, false);
            }
        }
        // scale + mask
#pragma unroll
        for (int s = 0; s < 2; ++s) {
            int t = t0 + s * 16 + m;
#pragma unroll
            for (int r = 0; r < 8; ++r) {
                int l = l0 + half * 8 + r;
                float v = S[s].f[r] * 0.125f;   // 1/sqrt(64)
                bool ok = true;
                if (kind == 1)      ok = (t <= l + off);
                else if (kind == 2) ok = (t < halfT) ? (t <= l) : ((t - halfT) <= l);
                S[s].f[r] = ok ? v : -3.0e38f;
            }
        }
        // online softmax update (row reductions within 16-lane halves)
#pragma unroll
        for (int r = 0; r < 8; ++r) {
            float tm = fmaxf(S[0].f[r], S[1].f[r]);
#pragma unroll
            for (int d = 8; d; d >>= 1) tm = fmaxf(tm, __shfl_xor(tm, d, 32));
            float mn = fmaxf(mrow[r], tm);
            float alpha = __expf(mrow[r] - mn);
            mrow[r] = mn;
            float p0 = __expf(S[0].f[r] - mn);
            float p1 = __expf(S[1].f[r] - mn);
            S[0].f[r] = p0; S[1].f[r] = p1;
            float rs = p0 + p1;
#pragma unroll
            for (int d = 8; d; d >>= 1) rs += __shfl_xor(rs, d, 32);
            lrow[r] = lrow[r] * alpha + rs;
#pragma unroll
            for (int ni = 0; ni < 4; ++ni) O[ni].f[r] *= alpha;
        }
        // route P through LDS: C-layout -> A-fragment layout
        __syncthreads();
#pragma unroll
        for (int s = 0; s < 2; ++s)
#pragma unroll
            for (int r = 0; r < 8; ++r)
                Plds[(half * 8 + r) * 32 + s * 16 + m] = f2bf(S[s].f[r]);
        __syncthreads();
        U16 ap;
        ap.q[0] = *(const uint4*)&Plds[m * 32 + half * 8];
        ap.q[1] = *(const uint4*)&Plds[m * 32 + 16 + half * 8];
        // O += P @ V  (contraction over 32 keys; 4 16-col value tiles)
        // V^T layout: contraction index t is contiguous -> two b128 loads
#pragma unroll
        for (int ni = 0; ni < 4; ++ni) {
            const unsigned short* vr = vbT + (size_t)(ni * 16 + m) * vld + t0 + half * 8;
            U16 bv_;
            bv_.q[0] = *(const uint4*)(vr);
            bv_.q[1] = *(const uint4*)(vr + 16);
            O[ni].v = __builtin_amdgcn_wmma_f32_16x16x32_bf16(
                false, ap.v, false, bv_.v, (short)0, O[ni].v, false, false);
        }
    }
    // normalize + store bf16
#pragma unroll
    for (int ni = 0; ni < 4; ++ni)
#pragma unroll
        for (int r = 0; r < 8; ++r) {
            int l = l0 + half * 8 + r;
            float val = O[ni].f[r] / lrow[r];
            Obf[((size_t)(b * Lq + l)) * 768 + h * 64 + ni * 16 + m] = f2bf(val);
        }
}

// ---------------------------------------------------------------------------
// Host-side orchestration
// ---------------------------------------------------------------------------
struct BlkP {
    const float *b1, *b2, *bk, *bo, *bq, *bv;
    const float *ln1_b, *ln1_g, *ln2_b, *ln2_g;
    const float *w1, *w2, *wk, *wo, *wq, *wv;
};

// Param leaves arrive in jax pytree flatten order (dict keys sorted):
// block dict: b1,b2,bk,bo,bq,bv,ln1_b,ln1_g,ln2_b,ln2_g,w1,w2,wk,wo,wq,wv
static BlkP get_blk(void* const* d_in, int base, int i) {
    BlkP p;
    p.b1    = (const float*)d_in[base + 0]  + (size_t)i * 3072;
    p.b2    = (const float*)d_in[base + 1]  + (size_t)i * 768;
    p.bk    = (const float*)d_in[base + 2]  + (size_t)i * 768;
    p.bo    = (const float*)d_in[base + 3]  + (size_t)i * 768;
    p.bq    = (const float*)d_in[base + 4]  + (size_t)i * 768;
    p.bv    = (const float*)d_in[base + 5]  + (size_t)i * 768;
    p.ln1_b = (const float*)d_in[base + 6]  + (size_t)i * 768;
    p.ln1_g = (const float*)d_in[base + 7]  + (size_t)i * 768;
    p.ln2_b = (const float*)d_in[base + 8]  + (size_t)i * 768;
    p.ln2_g = (const float*)d_in[base + 9]  + (size_t)i * 768;
    p.w1    = (const float*)d_in[base + 10] + (size_t)i * 768 * 3072;
    p.w2    = (const float*)d_in[base + 11] + (size_t)i * 3072 * 768;
    p.wk    = (const float*)d_in[base + 12] + (size_t)i * 768 * 768;
    p.wo    = (const float*)d_in[base + 13] + (size_t)i * 768 * 768;
    p.wq    = (const float*)d_in[base + 14] + (size_t)i * 768 * 768;
    p.wv    = (const float*)d_in[base + 15] + (size_t)i * 768 * 768;
    return p;
}

struct Scratch {
    unsigned short *ABF, *KVBF, *H1, *Qbf, *Kbf, *Vbf;
    float *X, *MEM;
};

static void mlp_tail(hipStream_t st, float* Xb, const BlkP& P, const Scratch& s) {
    ln_bf16_kernel<<<2048, 256, 0, st>>>(Xb, P.ln2_g, P.ln2_b, s.ABF);
    gemm_bf16_kernel<<<dim3(24, 16), 256, 0, st>>>(s.ABF, P.w1, P.b1, nullptr,
                                                   nullptr, s.H1, 2048, 3072, 768, 1, 0);
    gemm_bf16_kernel<<<dim3(6, 16), 256, 0, st>>>(s.H1, P.w2, P.b2, Xb,
                                                  Xb, nullptr, 2048, 768, 3072, 0, 0);
}

static void self_block(hipStream_t st, float* Xb, const BlkP& P, int kind, const Scratch& s) {
    dim3 g6(6, 16);
    ln_bf16_kernel<<<2048, 256, 0, st>>>(Xb, P.ln1_g, P.ln1_b, s.ABF);
    gemm_bf16_kernel<<<g6, 256, 0, st>>>(s.ABF, P.wq, P.bq, nullptr, nullptr, s.Qbf, 2048, 768, 768, 0, 0);
    gemm_bf16_kernel<<<g6, 256, 0, st>>>(s.ABF, P.wk, P.bk, nullptr, nullptr, s.Kbf, 2048, 768, 768, 0, 0);
    gemm_bf16_kernel<<<g6, 256, 0, st>>>(s.ABF, P.wv, P.bv, nullptr, nullptr, s.Vbf, 2048, 768, 768, 0, 1);
    flash_attn_kernel<<<dim3(64, 12, 2), 32, 0, st>>>(s.Qbf, s.Kbf, s.Vbf, s.ABF, 1024, 1024, 2048, kind);
    gemm_bf16_kernel<<<g6, 256, 0, st>>>(s.ABF, P.wo, P.bo, Xb, Xb, nullptr, 2048, 768, 768, 0, 0);
    mlp_tail(st, Xb, P, s);
}

// cross block: q/resid from Xq, kv = cat([part1, part2]) per batch, LN shared (ln1)
static void cross_block(hipStream_t st, float* Xq, const float* part1, const float* part2,
                        const BlkP& P, int kind, const Scratch& s) {
    dim3 g6(6, 16), g6kv(6, 32);
    ln_bf16_kernel<<<2048, 256, 0, st>>>(Xq, P.ln1_g, P.ln1_b, s.ABF);
    for (int b = 0; b < 2; ++b) {
        ln_bf16_kernel<<<1024, 256, 0, st>>>(part1 + (size_t)b * 1024 * 768, P.ln1_g, P.ln1_b,
                                             s.KVBF + (size_t)(b * 2048) * 768);
        ln_bf16_kernel<<<1024, 256, 0, st>>>(part2 + (size_t)b * 1024 * 768, P.ln1_g, P.ln1_b,
                                             s.KVBF + (size_t)(b * 2048 + 1024) * 768);
    }
    gemm_bf16_kernel<<<g6,   256, 0, st>>>(s.ABF,  P.wq, P.bq, nullptr, nullptr, s.Qbf, 2048, 768, 768, 0, 0);
    gemm_bf16_kernel<<<g6kv, 256, 0, st>>>(s.KVBF, P.wk, P.bk, nullptr, nullptr, s.Kbf, 4096, 768, 768, 0, 0);
    gemm_bf16_kernel<<<g6kv, 256, 0, st>>>(s.KVBF, P.wv, P.bv, nullptr, nullptr, s.Vbf, 4096, 768, 768, 0, 1);
    flash_attn_kernel<<<dim3(64, 12, 2), 32, 0, st>>>(s.Qbf, s.Kbf, s.Vbf, s.ABF, 1024, 2048, 4096, kind);
    gemm_bf16_kernel<<<g6, 256, 0, st>>>(s.ABF, P.wo, P.bo, Xq, Xq, nullptr, 2048, 768, 768, 0, 0);
    mlp_tail(st, Xq, P, s);
}

extern "C" void kernel_launch(void* const* d_in, const int* in_sizes, int n_in,
                              void* d_out, int out_size, void* d_ws, size_t ws_size,
                              hipStream_t stream) {
    (void)in_sizes; (void)n_in; (void)out_size; (void)ws_size;

    // Input order: ids, memory_in, then params leaves (dict keys sorted):
    // blocks_main(2..17), blocks_mc(18..33), blocks_mr(34..49),
    // lnf_b(50), lnf_g(51), w_head(52), wpe(53), wte(54)
    const int*   ids       = (const int*)d_in[0];
    const float* memory_in = (const float*)d_in[1];
    const float* lnf_b     = (const float*)d_in[50];
    const float* lnf_g     = (const float*)d_in[51];
    const float* w_head    = (const float*)d_in[52];
    const float* wpe       = (const float*)d_in[53];
    const float* wte       = (const float*)d_in[54];

    // workspace carve-out
    char* p = (char*)d_ws;
    auto carve = [&](size_t bytes) { char* q = p; p += (bytes + 255) & ~(size_t)255; return q; };
    Scratch s;
    s.ABF  = (unsigned short*)carve((size_t)2048 * 768 * 2);
    s.KVBF = (unsigned short*)carve((size_t)4096 * 768 * 2);
    s.H1   = (unsigned short*)carve((size_t)2048 * 3072 * 2);
    s.Qbf  = (unsigned short*)carve((size_t)2048 * 768 * 2);
    s.Kbf  = (unsigned short*)carve((size_t)4096 * 768 * 2);
    s.Vbf  = (unsigned short*)carve((size_t)4096 * 768 * 2);   // stored transposed
    s.X    = (float*)carve((size_t)2048 * 768 * 4);
    s.MEM  = (float*)carve((size_t)2048 * 768 * 4);

    float* logits = (float*)d_out;
    float* MC     = (float*)d_out + (size_t)2048 * 50257;   // mem_created region

    // 1. embedding
    embed_kernel<<<(2048 * 768 + 255) / 256, 256, 0, stream>>>(ids, wte, wpe, s.X);

    // 2. blocks1 (main 0,1) causal
    self_block(stream, s.X, get_blk(d_in, 2, 0), 1, s);
    self_block(stream, s.X, get_blk(d_in, 2, 1), 1, s);

    // 3. mem_ret = x; blocks_mr1 (mr 0,1) causal
    hipMemcpyAsync(s.MEM, s.X, (size_t)2048 * 768 * 4, hipMemcpyDeviceToDevice, stream);
    self_block(stream, s.MEM, get_blk(d_in, 34, 0), 1, s);
    self_block(stream, s.MEM, get_blk(d_in, 34, 1), 1, s);

    // 4. mem_created = mem_ret; mc blocks with 'full' mask (in-place in d_out)
    hipMemcpyAsync(MC, s.MEM, (size_t)2048 * 768 * 4, hipMemcpyDeviceToDevice, stream);
    self_block(stream, MC, get_blk(d_in, 18, 0), 0, s);
    self_block(stream, MC, get_blk(d_in, 18, 1), 0, s);

    // 5. mr2: cross-attn over cat([mem_ret, memory_in]), causal (offset 1024)
    cross_block(stream, s.MEM, s.MEM, memory_in, get_blk(d_in, 34, 2), 1, s);
    self_block(stream, s.MEM, get_blk(d_in, 34, 3), 1, s);

    // 6. blocks2 (main 2,3) causal
    self_block(stream, s.X, get_blk(d_in, 2, 2), 1, s);
    self_block(stream, s.X, get_blk(d_in, 2, 3), 1, s);

    // 7. blocks3: doublecausal cross over cat([mem_ret, x]), then causal
    cross_block(stream, s.X, s.MEM, s.X, get_blk(d_in, 2, 4), 2, s);
    self_block(stream, s.X, get_blk(d_in, 2, 5), 1, s);

    // 8. final LN + head
    ln_bf16_kernel<<<2048, 256, 0, stream>>>(s.X, lnf_g, lnf_b, s.ABF);
    gemm_bf16_kernel<<<dim3((50257 + 127) / 128, 16), 256, 0, stream>>>(
        s.ABF, w_head, nullptr, nullptr, logits, nullptr, 2048, 50257, 768, 0, 0);
}
